// DeformableTransformer_67430986547700
// MI455X (gfx1250) — matrix operand order repaired and checked by reading, not compiled, not run on hardware
//
#include <hip/hip_runtime.h>
#include <math.h>

// ---------------------------------------------------------------------------
// MI455X (gfx1250) implementation of a 6-layer deformable-DETR decoder.
// All GEMM contractions run on v_wmma_f32_16x16x32_bf16 (bf16 in, f32 acc).
// GEMM weight tiles are staged to LDS with CDNA5 async DMA
// (global_load_async_to_lds_b128 + s_wait_asynccnt), double-buffered.
// Attention is fused flash-style (online softmax, scores never hit HBM).
// ---------------------------------------------------------------------------

typedef __attribute__((ext_vector_type(16))) __bf16 v16bf;
typedef __attribute__((ext_vector_type(8)))  __bf16 v8bf;
typedef __attribute__((ext_vector_type(8)))  float  v8f;

#define WMMA_BF16(a, b, c) \
  __builtin_amdgcn_wmma_f32_16x16x32_bf16(false, (a), false, (b), (short)0, (c), false, false)

constexpr int Bc = 4, Qc = 900, Dc = 256, Hc = 8, DHc = 32;
constexpr int Sc = 19947;          // sum of level sizes
constexpr int Qp = 928;            // Q padded to mult of 32 (928*2B = 58*32B, keeps v16bf aligned)
constexpr int BQ = Bc * Qc;        // 3600
constexpr int BS = Bc * Sc;        // 79788

// ---------------------------------------------------------------------------
// Generic WMMA GEMM:  Out[M,N] = epilogue(A[M,K] @ W[N,K]^T + bias)
// Block = 256 threads = 8 waves; tile 128(M) x 64(N); each wave owns 16x64.
// B (weight) 64x32 tile staged to LDS via async DMA, shared by all 8 waves
// (kills the 8x redundant global loads); A frag loaded per-wave from global.
// A and B fragments both use contiguous k order: k = 16*(lane/16) + s.
// ---------------------------------------------------------------------------
template <int EPI /*0=bias,1=bias+relu*/, bool OB /*bf16 out*/>
__global__ __launch_bounds__(256) void k_gemm(
    const __bf16* __restrict__ A, const __bf16* __restrict__ W,
    const float* __restrict__ bias, float* __restrict__ outF,
    __bf16* __restrict__ outB, int M, int N, int K) {
  __shared__ __attribute__((aligned(64))) __bf16 bt[2][64][32];  // [buf][n][k]

  const int tid  = threadIdx.x;
  const int wave = tid >> 5;
  const int lane = tid & 31;
  const int ln15 = lane & 15;
  const int hi   = lane >> 4;           // half-wave select
  const int m0 = blockIdx.x * 128 + wave * 16;
  const int n0 = blockIdx.y * 64;

  int arow = m0 + ln15;                 // A row for this lane (clamp for ragged M)
  if (arow > M - 1) arow = M - 1;
  const __bf16* aptr = A + (size_t)arow * K;

  // async copy descriptors: thread -> 8 contiguous bf16 (16B) of the B tile
  const int brow = tid >> 2;            // 0..63 (tile column n)
  const int bk   = (tid & 3) * 8;       // 0,8,16,24 (k offset)
  const __bf16* wsrc = W + (size_t)(n0 + brow) * K + bk;

  v8f acc[4];
#pragma unroll
  for (int t = 0; t < 4; ++t)
#pragma unroll
    for (int i = 0; i < 8; ++i) acc[t][i] = 0.f;

  // issue tile 0
  {
    uint32_t lo = (uint32_t)(uintptr_t)(&bt[0][brow][bk]);
    uint64_t ga = (uint64_t)(uintptr_t)(wsrc);
    asm volatile("global_load_async_to_lds_b128 %0, %1, off"
                 :: "v"(lo), "v"(ga) : "memory");
  }

  const int nk = K >> 5;
  for (int it = 0; it < nk; ++it) {
    const int k0  = it << 5;
    const int cur = it & 1;
    if (it + 1 < nk) {                  // prefetch next tile into the other buffer
      uint32_t lo = (uint32_t)(uintptr_t)(&bt[cur ^ 1][brow][bk]);
      uint64_t ga = (uint64_t)(uintptr_t)(wsrc + k0 + 32);
      asm volatile("global_load_async_to_lds_b128 %0, %1, off"
                   :: "v"(lo), "v"(ga) : "memory");
      asm volatile("s_wait_asynccnt 1" ::: "memory");   // tile 'it' landed
    } else {
      asm volatile("s_wait_asynccnt 0" ::: "memory");
    }
    __syncthreads();                    // tile visible to all waves

    union { v16bf v; v8bf h[2]; } a;
    // lanes 0-15: K = k0+{0..15}; lanes 16-31: +16  -> base = k0 + hi*16
    a.h[0] = *(const v8bf*)(aptr + k0 + hi * 16);
    a.h[1] = *(const v8bf*)(aptr + k0 + hi * 16 + 8);
    __builtin_prefetch(aptr + k0 + 64, 0, 1);   // global_prefetch_b8
#pragma unroll
    for (int t = 0; t < 4; ++t) {
      // B frag from LDS: lane n = t*16+ln15, K = hi*16 + j (contiguous 16)
      v16bf b = *(const v16bf*)(&bt[cur][t * 16 + ln15][hi * 16]);
      acc[t] = WMMA_BF16(a.v, b, acc[t]);
    }
    __syncthreads();                    // reads done; buffer reusable
  }

#pragma unroll
  for (int t = 0; t < 4; ++t) {
    const int col = n0 + t * 16 + ln15;
    const float bv = bias ? bias[col] : 0.f;
#pragma unroll
    for (int r = 0; r < 8; ++r) {
      const int row = m0 + r + hi * 8;  // f32 C layout: vgpr r -> row r (+8 hi half)
      if (row < M) {
        float v = acc[t][r] + bv;
        if (EPI == 1) v = fmaxf(v, 0.f);
        if (OB) outB[(size_t)row * N + col] = (__bf16)v;
        else    outF[(size_t)row * N + col] = v;
      }
    }
  }
}

// ---------------------------------------------------------------------------
// Fused flash attention: one block per (b*h, 128-row Q tile); 8 waves x 16 rows.
// qh/kh: bf16 [B*H, Qp, 32] (zero padded); vhT: bf16 [B*H, 32, Qp] (transposed).
// Online softmax with half-wave shfl_xor row reductions; P staged via LDS.
// ---------------------------------------------------------------------------
__global__ __launch_bounds__(256) void k_attn(
    const __bf16* __restrict__ qh, const __bf16* __restrict__ kh,
    const __bf16* __restrict__ vhT, float* __restrict__ out) {
  __shared__ __attribute__((aligned(64))) __bf16 plds[8][16][32];
  const int wave = threadIdx.x >> 5;
  const int lane = threadIdx.x & 31;
  const int ln15 = lane & 15;
  const int hi   = lane >> 4;
  const int bh = blockIdx.x;            // 0..31
  const int b = bh >> 3, h = bh & 7;
  const int q0 = blockIdx.y * 128 + wave * 16;

  int arow = q0 + ln15;
  if (arow > Qp - 1) arow = Qp - 1;     // padded rows are zero; stores guarded below
  const __bf16* qp = qh + ((size_t)bh * Qp + arow) * DHc + hi * 16;
  union { v16bf v; v8bf h2[2]; } qa;
  qa.h2[0] = *(const v8bf*)qp;
  qa.h2[1] = *(const v8bf*)(qp + 8);

  v8f o0, o1, zf;
  float m[8], l[8];
#pragma unroll
  for (int r = 0; r < 8; ++r) { o0[r] = 0.f; o1[r] = 0.f; zf[r] = 0.f; m[r] = -1e30f; l[r] = 0.f; }

  const float scale = 0.17677669529663687f;   // 1/sqrt(32)

  for (int kc = 0; kc < Qp; kc += 32) {
    // ---- scores: 16 rows x 32 kv cols = two WMMAs -------------------------
    v16bf bk0 = *(const v16bf*)(kh + ((size_t)bh * Qp + kc + ln15) * DHc + hi * 16);
    v16bf bk1 = *(const v16bf*)(kh + ((size_t)bh * Qp + kc + 16 + ln15) * DHc + hi * 16);
    v8f s0 = WMMA_BF16(qa.v, bk0, zf);
    v8f s1 = WMMA_BF16(qa.v, bk1, zf);
    const bool c0ok = (kc + ln15) < Qc;
    const bool c1ok = (kc + 16 + ln15) < Qc;
#pragma unroll
    for (int r = 0; r < 8; ++r) {
      float v0 = c0ok ? s0[r] * scale : -1e30f;
      float v1 = c1ok ? s1[r] * scale : -1e30f;
      float mx = fmaxf(v0, v1);
#pragma unroll
      for (int o = 1; o < 16; o <<= 1) mx = fmaxf(mx, __shfl_xor(mx, o, 32));
      const float mn = fmaxf(m[r], mx);
      const float f  = __expf(m[r] - mn);
      const float p0 = __expf(v0 - mn);
      const float p1 = __expf(v1 - mn);
      float sum = p0 + p1;
#pragma unroll
      for (int o = 1; o < 16; o <<= 1) sum += __shfl_xor(sum, o, 32);
      m[r] = mn;
      l[r] = l[r] * f + sum;
      o0[r] *= f; o1[r] *= f;
      plds[wave][r + hi * 8][ln15]      = (__bf16)p0;
      plds[wave][r + hi * 8][16 + ln15] = (__bf16)p1;
    }
    asm volatile("s_wait_dscnt 0" ::: "memory");   // cross-lane LDS RAW fence
    // ---- P (A-frag from LDS) @ V -----------------------------------------
    union { v16bf v; v8bf h2[2]; } pa;
    const __bf16* pp = &plds[wave][ln15][hi * 16];
    pa.h2[0] = *(const v8bf*)pp;
    pa.h2[1] = *(const v8bf*)(pp + 8);
    v16bf bv0 = *(const v16bf*)(vhT + ((size_t)bh * DHc + ln15) * Qp + kc + hi * 16);
    v16bf bv1 = *(const v16bf*)(vhT + ((size_t)bh * DHc + 16 + ln15) * Qp + kc + hi * 16);
    o0 = WMMA_BF16(pa.v, bv0, o0);
    o1 = WMMA_BF16(pa.v, bv1, o1);
  }

#pragma unroll
  for (int r = 0; r < 8; ++r) {
    const int q = q0 + r + hi * 8;
    if (q < Qc) {
      const float inv = 1.f / l[r];
      out[((size_t)b * Qc + q) * Dc + h * DHc + ln15]      = o0[r] * inv;
      out[((size_t)b * Qc + q) * Dc + h * DHc + 16 + ln15] = o1[r] * inv;
    }
  }
}

// ---------------------------------------------------------------------------
// Small helper kernels
// ---------------------------------------------------------------------------
__global__ void k_cvt(const float* __restrict__ s, __bf16* __restrict__ d, int n) {
  int i = blockIdx.x * 256 + threadIdx.x;
  if (i < n) d[i] = (__bf16)s[i];
}
__global__ void k_addcvt(const float* __restrict__ a, const float* __restrict__ b,
                         __bf16* __restrict__ d, int n) {
  int i = blockIdx.x * 256 + threadIdx.x;
  if (i < n) d[i] = (__bf16)(a[i] + b[i]);
}

// qpos = pos2posemb(reference_points): out[c*64 + 2j] = sin, [.. +1] = cos
__global__ void k_posemb(const float* __restrict__ ref, float* __restrict__ qpos) {
  int idx = blockIdx.x * 256 + threadIdx.x;
  if (idx >= BQ * Dc) return;
  const int d = idx & 255, bq = idx >> 8;
  const int c = d >> 6, i = d & 63, j = i >> 1;
  const float pos = ref[bq * 4 + c] * 6.283185307179586f;
  const float a = pos / powf(10000.f, (float)j * (1.f / 32.f));
  qpos[idx] = (i & 1) ? cosf(a) : sinf(a);
}

// qk f32 [BQ,512] -> qh/kh bf16 [B*H, Qp, 32] (zero pad rows >= Q)
__global__ void k_split_qk(const float* __restrict__ qk, __bf16* __restrict__ qh,
                           __bf16* __restrict__ kh) {
  int idx = blockIdx.x * 256 + threadIdx.x;
  if (idx >= Bc * Hc * Qp * DHc) return;
  const int dh = idx & 31;
  int t = idx >> 5;
  const int qi = t % Qp; t /= Qp;
  const int h = t % Hc, b = t / Hc;
  __bf16 qv = (__bf16)0.f, kv = (__bf16)0.f;
  if (qi < Qc) {
    const float* row = qk + ((size_t)b * Qc + qi) * 512;
    qv = (__bf16)row[h * 32 + dh];
    kv = (__bf16)row[256 + h * 32 + dh];
  }
  qh[idx] = qv;
  kh[idx] = kv;
}

// v f32 [BQ,256] -> vhT bf16 [B*H, 32, Qp] transposed (zero pad)
__global__ void k_split_vT(const float* __restrict__ v, __bf16* __restrict__ vhT) {
  int idx = blockIdx.x * 256 + threadIdx.x;
  if (idx >= Bc * Hc * DHc * Qp) return;
  const int qi = idx % Qp;
  int t = idx / Qp;
  const int dh = t % 32; t /= 32;
  const int h = t % Hc, b = t / Hc;
  vhT[idx] = (qi < Qc) ? (__bf16)v[((size_t)b * Qc + qi) * Dc + h * 32 + dh] : (__bf16)0.f;
}

// out = layernorm(x + r) * g + b ; one block (256 thr) per row, D=256
__global__ __launch_bounds__(256) void k_add_ln(
    const float* __restrict__ x, const float* __restrict__ r,
    const float* __restrict__ g, const float* __restrict__ bb,
    float* __restrict__ out) {
  __shared__ float red[8];
  const int row = blockIdx.x, t = threadIdx.x;
  const int w = t >> 5, ln = t & 31;
  const float v = x[(size_t)row * 256 + t] + r[(size_t)row * 256 + t];
  float s = v;
#pragma unroll
  for (int o = 16; o > 0; o >>= 1) s += __shfl_xor(s, o, 32);
  if (ln == 0) red[w] = s;
  __syncthreads();
  float tot = 0.f;
#pragma unroll
  for (int i = 0; i < 8; ++i) tot += red[i];
  const float mean = tot * (1.f / 256.f);
  __syncthreads();
  const float dv = v - mean;
  float sq = dv * dv;
#pragma unroll
  for (int o = 16; o > 0; o >>= 1) sq += __shfl_xor(sq, o, 32);
  if (ln == 0) red[w] = sq;
  __syncthreads();
  float tv = 0.f;
#pragma unroll
  for (int i = 0; i < 8; ++i) tv += red[i];
  const float inv = rsqrtf(tv * (1.f / 256.f) + 1e-5f);
  out[(size_t)row * 256 + t] = dv * inv * g[t] + bb[t];
}

// softmax over 16 (L*P) per (b,q,h), in place on [BQ,128]
__global__ void k_aw_softmax(float* __restrict__ aw) {
  int i = blockIdx.x * 256 + threadIdx.x;
  if (i >= BQ * Hc) return;
  const int h = i % Hc, bq = i / Hc;
  float* p = aw + (size_t)bq * 128 + h * 16;
  float mx = -1e30f;
#pragma unroll
  for (int j = 0; j < 16; ++j) mx = fmaxf(mx, p[j]);
  float e[16], s = 0.f;
#pragma unroll
  for (int j = 0; j < 16; ++j) { e[j] = __expf(p[j] - mx); s += e[j]; }
  const float inv = 1.f / s;
#pragma unroll
  for (int j = 0; j < 16; ++j) p[j] = e[j] * inv;
}

// Multi-scale deformable sampling: thread per (b,q,d); 16 bilinear gathers.
__global__ __launch_bounds__(256) void k_msda(
    const float* __restrict__ ref, const float* __restrict__ vr,
    const float* __restrict__ offb, const float* __restrict__ aw,
    const __bf16* __restrict__ val, float* __restrict__ out) {
  const int idx = blockIdx.x * 256 + threadIdx.x;
  if (idx >= BQ * Dc) return;
  const int d = idx & 255;
  const int bq = idx >> 8;
  const int b = bq / Qc;
  const int h = d >> 5;
  const float r0 = ref[bq * 4], r1 = ref[bq * 4 + 1];
  const float r2 = ref[bq * 4 + 2], r3 = ref[bq * 4 + 3];
  const float* orow = offb + (size_t)bq * 256 + h * 32;
  const float* arow = aw + (size_t)bq * 128 + h * 16;
  const int HLs[4] = {100, 50, 25, 13};
  const int WLs[4] = {150, 75, 38, 19};
  const int S0s[4] = {0, 15000, 18750, 19700};
  float acc = 0.f;
#pragma unroll
  for (int l = 0; l < 4; ++l) {
    const float vx = vr[(b * 4 + l) * 2], vy = vr[(b * 4 + l) * 2 + 1];
    const float cx = r0 * vx, cy = r1 * vy, wv = r2 * vx, hv = r3 * vy;
    const int Hl = HLs[l], Wl = WLs[l], s0 = S0s[l];
#pragma unroll
    for (int p = 0; p < 4; ++p) {
      const float lx = cx + orow[l * 8 + p * 2]     * 0.25f * wv * 0.5f;
      const float ly = cy + orow[l * 8 + p * 2 + 1] * 0.25f * hv * 0.5f;
      const float xf = lx * Wl - 0.5f, yf = ly * Hl - 0.5f;
      const float x0 = floorf(xf), y0 = floorf(yf);
      const int ix = (int)x0, iy = (int)y0;
      const float wx = xf - x0, wy = yf - y0;
      float s00 = 0.f, s10 = 0.f, s01 = 0.f, s11 = 0.f;
      if (iy >= 0 && iy < Hl) {
        const size_t rb = ((size_t)b * Sc + s0 + (size_t)iy * Wl) * 256 + d;
        if (ix >= 0 && ix < Wl)         s00 = (float)val[rb + (size_t)ix * 256];
        if (ix + 1 >= 0 && ix + 1 < Wl) s10 = (float)val[rb + (size_t)(ix + 1) * 256];
      }
      if (iy + 1 >= 0 && iy + 1 < Hl) {
        const size_t rb = ((size_t)b * Sc + s0 + (size_t)(iy + 1) * Wl) * 256 + d;
        if (ix >= 0 && ix < Wl)         s01 = (float)val[rb + (size_t)ix * 256];
        if (ix + 1 >= 0 && ix + 1 < Wl) s11 = (float)val[rb + (size_t)(ix + 1) * 256];
      }
      const float bil = s00 * (1.f - wx) * (1.f - wy) + s10 * wx * (1.f - wy)
                      + s01 * (1.f - wx) * wy + s11 * wx * wy;
      acc += arow[l * 4 + p] * bil;
    }
  }
  out[idx] = acc;
}

// ---------------------------------------------------------------------------
// Orchestration
// ---------------------------------------------------------------------------
extern "C" void kernel_launch(void* const* d_in, const int* in_sizes, int n_in,
                              void* d_out, int out_size, void* d_ws, size_t ws_size,
                              hipStream_t stream) {
  (void)in_sizes; (void)n_in; (void)out_size; (void)ws_size;
  const float* tgt = (const float*)d_in[0];
  const float* ref = (const float*)d_in[1];
  const float* src = (const float*)d_in[2];
  const float* vr  = (const float*)d_in[3];
  const float* msda_off_w = (const float*)d_in[4];
  const float* msda_off_b = (const float*)d_in[5];
  const float* msda_aw_w  = (const float*)d_in[6];
  const float* msda_aw_b  = (const float*)d_in[7];
  const float* msda_val_w = (const float*)d_in[8];
  const float* msda_val_b = (const float*)d_in[9];
  const float* msda_out_w = (const float*)d_in[10];
  const float* msda_out_b = (const float*)d_in[11];
  const float* sa_in_w  = (const float*)d_in[12];
  const float* sa_in_b  = (const float*)d_in[13];
  const float* sa_out_w = (const float*)d_in[14];
  const float* sa_out_b = (const float*)d_in[15];
  const float* ffn_w1 = (const float*)d_in[16];
  const float* ffn_b1 = (const float*)d_in[17];
  const float* ffn_w2 = (const float*)d_in[18];
  const float* ffn_b2 = (const float*)d_in[19];
  const float* ln_g = (const float*)d_in[20];
  const float* ln_b = (const float*)d_in[21];

  char* ws = (char*)d_ws;
  size_t woff = 0;
  auto alloc = [&](size_t bytes) -> char* {
    char* p = ws + woff;
    woff = (woff + bytes + 255) & ~(size_t)255;
    return p;
  };
  float*  qpos    = (float*)alloc((size_t)BQ * Dc * 4);
  float*  xA      = (float*)alloc((size_t)BQ * Dc * 4);
  float*  xB      = (float*)alloc((size_t)BQ * Dc * 4);
  float*  ybuf    = (float*)alloc((size_t)BQ * Dc * 4);
  float*  qk      = (float*)alloc((size_t)BQ * 512 * 4);
  float*  vout    = (float*)alloc((size_t)BQ * Dc * 4);
  float*  attn    = (float*)alloc((size_t)BQ * Dc * 4);
  float*  offbuf  = (float*)alloc((size_t)BQ * Dc * 4);
  float*  awlin   = (float*)alloc((size_t)BQ * 128 * 4);
  float*  crossp  = (float*)alloc((size_t)BQ * Dc * 4);
  __bf16* act0    = (__bf16*)alloc((size_t)BQ * Dc * 2);
  __bf16* act1    = (__bf16*)alloc((size_t)BQ * Dc * 2);
  __bf16* wbf     = (__bf16*)alloc((size_t)1024 * 256 * 2);
  __bf16* qhB     = (__bf16*)alloc((size_t)Bc * Hc * Qp * DHc * 2);
  __bf16* khB     = (__bf16*)alloc((size_t)Bc * Hc * Qp * DHc * 2);
  __bf16* vhT     = (__bf16*)alloc((size_t)Bc * Hc * DHc * Qp * 2);
  __bf16* srcbf   = (__bf16*)alloc((size_t)BS * Dc * 2);
  __bf16* valuebf = (__bf16*)alloc((size_t)BS * Dc * 2);
  __bf16* ffnh    = (__bf16*)alloc((size_t)BQ * 1024 * 2);

  auto cvt = [&](const float* s, __bf16* d, int n) {
    k_cvt<<<(n + 255) / 256, 256, 0, stream>>>(s, d, n);
  };
  auto gemmF = [&](const __bf16* A, const __bf16* W, const float* bias, float* o,
                   int M, int N, int K) {
    dim3 g((M + 127) / 128, N / 64);
    k_gemm<0, false><<<g, 256, 0, stream>>>(A, W, bias, o, nullptr, M, N, K);
  };
  auto gemmB = [&](const __bf16* A, const __bf16* W, const float* bias, __bf16* o,
                   int M, int N, int K, bool relu) {
    dim3 g((M + 127) / 128, N / 64);
    if (relu) k_gemm<1, true><<<g, 256, 0, stream>>>(A, W, bias, nullptr, o, M, N, K);
    else      k_gemm<0, true><<<g, 256, 0, stream>>>(A, W, bias, nullptr, o, M, N, K);
  };

  // one-time prep
  k_posemb<<<(BQ * Dc + 255) / 256, 256, 0, stream>>>(ref, qpos);
  cvt(src, srcbf, BS * Dc);
  hipMemcpyAsync(xA, tgt, (size_t)BQ * Dc * 4, hipMemcpyDeviceToDevice, stream);

  float* x = xA;
  float* xn = xB;
  const int splitN = Bc * Hc * Qp * DHc;

  for (int li = 0; li < 6; ++li) {
    // ---------------- self attention (ln index 1) ----------------
    k_addcvt<<<(BQ * Dc + 255) / 256, 256, 0, stream>>>(x, qpos, act0, BQ * Dc);  // q = x+qpos
    cvt(x, act1, BQ * Dc);                                                        // v input = x
    cvt(sa_in_w + (size_t)li * 768 * 256, wbf, 768 * 256);
    gemmF(act0, wbf, sa_in_b + (size_t)li * 768, qk, BQ, 512, 256);               // q,k proj
    gemmF(act1, wbf + (size_t)512 * 256, sa_in_b + (size_t)li * 768 + 512, vout,
          BQ, 256, 256);                                                          // v proj
    k_split_qk<<<(splitN + 255) / 256, 256, 0, stream>>>(qk, qhB, khB);
    k_split_vT<<<(splitN + 255) / 256, 256, 0, stream>>>(vout, vhT);
    k_attn<<<dim3(Bc * Hc, 8), 256, 0, stream>>>(qhB, khB, vhT, attn);
    cvt(attn, act0, BQ * Dc);
    cvt(sa_out_w + (size_t)li * 256 * 256, wbf, 256 * 256);
    gemmF(act0, wbf, sa_out_b + (size_t)li * 256, ybuf, BQ, 256, 256);
    k_add_ln<<<BQ, 256, 0, stream>>>(x, ybuf, ln_g + (size_t)(li * 3 + 1) * 256,
                                     ln_b + (size_t)(li * 3 + 1) * 256, xn);
    { float* t = x; x = xn; xn = t; }

    // ---------------- deformable cross attention (ln index 0) ----------------
    k_addcvt<<<(BQ * Dc + 255) / 256, 256, 0, stream>>>(x, qpos, act0, BQ * Dc);  // query
    cvt(msda_off_w + (size_t)li * 256 * 256, wbf, 256 * 256);
    gemmF(act0, wbf, msda_off_b + (size_t)li * 256, offbuf, BQ, 256, 256);
    cvt(msda_aw_w + (size_t)li * 128 * 256, wbf, 128 * 256);
    gemmF(act0, wbf, msda_aw_b + (size_t)li * 128, awlin, BQ, 128, 256);
    k_aw_softmax<<<(BQ * Hc + 255) / 256, 256, 0, stream>>>(awlin);
    cvt(msda_val_w + (size_t)li * 256 * 256, wbf, 256 * 256);
    gemmB(srcbf, wbf, msda_val_b + (size_t)li * 256, valuebf, BS, 256, 256, false);
    k_msda<<<(BQ * Dc + 255) / 256, 256, 0, stream>>>(ref, vr, offbuf, awlin,
                                                      valuebf, crossp);
    cvt(crossp, act0, BQ * Dc);
    cvt(msda_out_w + (size_t)li * 256 * 256, wbf, 256 * 256);
    gemmF(act0, wbf, msda_out_b + (size_t)li * 256, ybuf, BQ, 256, 256);
    k_add_ln<<<BQ, 256, 0, stream>>>(x, ybuf, ln_g + (size_t)(li * 3 + 0) * 256,
                                     ln_b + (size_t)(li * 3 + 0) * 256, xn);
    { float* t = x; x = xn; xn = t; }

    // ---------------- FFN (ln index 2) ----------------
    cvt(x, act0, BQ * Dc);
    cvt(ffn_w1 + (size_t)li * 1024 * 256, wbf, 1024 * 256);
    gemmB(act0, wbf, ffn_b1 + (size_t)li * 1024, ffnh, BQ, 1024, 256, true);      // relu
    cvt(ffn_w2 + (size_t)li * 256 * 1024, wbf, 256 * 1024);
    gemmF(ffnh, wbf, ffn_b2 + (size_t)li * 256, ybuf, BQ, 256, 1024);
    k_add_ln<<<BQ, 256, 0, stream>>>(x, ybuf, ln_g + (size_t)(li * 3 + 2) * 256,
                                     ln_b + (size_t)(li * 3 + 2) * 256, xn);
    { float* t = x; x = xn; xn = t; }
  }

  hipMemcpyAsync(d_out, x, (size_t)BQ * Dc * 4, hipMemcpyDeviceToDevice, stream);
}
